// MultiHeadAttention_18562848653755
// MI455X (gfx1250) — compile-verified, matrix operand order checked by reading, not run on hardware
//
#include <hip/hip_runtime.h>

// ---------------------------------------------------------------------------
// MultiHeadAttention forward for MI455X (gfx1250, wave32, WMMA).
// Pipeline: fp32->bf16 convert, fused QKV GEMM + bias + RoPE (WMMA bf16,
// fp32 accum, 32x64 per-wave blocking), flash-attention (WMMA bf16 for
// QK^T and PV, online softmax, LDS C->A relayout).
// ---------------------------------------------------------------------------

typedef __attribute__((ext_vector_type(16))) __bf16 v16bf;
typedef __attribute__((ext_vector_type(8)))  float  v8f;

#define S_LEN  2048
#define E_DIM  1024
#define H_NUM  16
#define D_DIM  64
#define B_NUM  2
#define M_ROWS (B_NUM * S_LEN)   // 4096

union FragU { v16bf v; uint4 q[2]; };

__device__ __forceinline__ unsigned short f32_bf16(float f) {
  unsigned int u = __float_as_uint(f);
  u += 0x7FFFu + ((u >> 16) & 1u);          // round-to-nearest-even
  return (unsigned short)(u >> 16);
}

// Loads a 16x32 bf16 fragment in the CDNA5 A/B VGPR striping from a
// row-major bf16 array (lane L%16 picks row/col, lane/16 picks K-half).
// Two contiguous 16-byte loads per lane.
__device__ __forceinline__ v16bf load_frag(const unsigned short* __restrict__ base,
                                           int stride, int row0, int k0, int lane) {
  int half = lane >> 4, lm = lane & 15;
  const unsigned short* p =
      base + (size_t)(row0 + lm) * (size_t)stride + (size_t)(k0 + 8 * half);
  FragU f;
  f.q[0] = *(const uint4*)(p);
  f.q[1] = *(const uint4*)(p + 16);
  return f.v;
}

__device__ __forceinline__ v8f wmma_bf16(v16bf a, v16bf b, v8f c) {
  // (neg_a, A, neg_b, B, c_mod, C, reuse_a, reuse_b)
  return __builtin_amdgcn_wmma_f32_16x16x32_bf16(false, a, false, b,
                                                 (short)0, c, false, false);
}

// ---------------- kernel 1: x fp32 -> bf16 ---------------------------------
__global__ void k_cvt_x(const float4* __restrict__ x, ushort4* __restrict__ xb, int n4) {
  int i = blockIdx.x * blockDim.x + threadIdx.x;
  if (i >= n4) return;
  float4 v = x[i];
  ushort4 r;
  r.x = f32_bf16(v.x); r.y = f32_bf16(v.y);
  r.z = f32_bf16(v.z); r.w = f32_bf16(v.w);
  xb[i] = r;
}

// ---------------- kernel 2: W fp32 (K,N) -> bf16 transposed (N,K) ----------
__global__ void k_cvt_wt(const float* __restrict__ Wq, const float* __restrict__ Wk,
                         const float* __restrict__ Wv, unsigned short* __restrict__ wt) {
  int idx = blockIdx.x * blockDim.x + threadIdx.x;   // 0 .. 3*2^20-1
  int w   = idx >> 20;
  int rem = idx & ((1 << 20) - 1);
  int n   = rem >> 10;
  int k   = rem & 1023;
  const float* W = (w == 0) ? Wq : ((w == 1) ? Wk : Wv);
  wt[((size_t)w << 20) + ((size_t)n << 10) + k] = f32_bf16(W[((size_t)k << 10) + n]);
}

// ---------------- kernel 3: fused QKV GEMM + bias + RoPE -------------------
// One wave -> 32 rows x 64 cols (= one full head) of one of Q/K/V:
// 2 A-fragments x 4 B-fragments = 8 WMMAs per 32-deep K step (B reused 2x).
// Q,K written bf16 in (B,H,S,D); V written bf16 transposed (B,H,D,S).
__global__ void __launch_bounds__(256)
k_qkv(const unsigned short* __restrict__ xb, const unsigned short* __restrict__ wt,
      const float* __restrict__ bq, const float* __restrict__ bk,
      const float* __restrict__ bv,
      unsigned short* __restrict__ qb, unsigned short* __restrict__ kb,
      unsigned short* __restrict__ vb) {
  int lane = threadIdx.x & 31;
  int widG = blockIdx.x * 8 + (threadIdx.x >> 5);
  int mat  = widG >> 11;          // 0=Q 1=K 2=V  (2048 waves per matrix)
  int rem  = widG & 2047;
  int m0   = (rem >> 4) << 5;     // 32-row tile (0..4064)
  int n0   = (rem & 15) << 6;     // 64-col tile (one head)

  const unsigned short* wb = wt + ((size_t)mat << 20);

  v8f acc[2][4];
#pragma unroll
  for (int mi = 0; mi < 2; ++mi)
#pragma unroll
    for (int t = 0; t < 4; ++t) acc[mi][t] = (v8f)(0.0f);

  for (int k0 = 0; k0 < E_DIM; k0 += 32) {
    v16bf a0 = load_frag(xb, E_DIM, m0,      k0, lane);
    v16bf a1 = load_frag(xb, E_DIM, m0 + 16, k0, lane);
    // Unconditional stream hint (worst case reads 64B into adjacent ws
    // region, still mapped; prefetch is speculative and side-effect free).
    __builtin_prefetch(xb + (size_t)(m0 + (lane & 15)) * E_DIM + k0 + 32, 0, 1);
#pragma unroll
    for (int t = 0; t < 4; ++t) {
      v16bf b = load_frag(wb, E_DIM, n0 + 16 * t, k0, lane);
      acc[0][t] = wmma_bf16(a0, b, acc[0][t]);
      acc[1][t] = wmma_bf16(a1, b, acc[1][t]);
    }
  }

  int half = lane >> 4, lm = lane & 15;
  const float* bias = (mat == 0) ? bq : ((mat == 1) ? bk : bv);
#pragma unroll
  for (int t = 0; t < 4; ++t) {
    float bval = bias[n0 + 16 * t + lm];
#pragma unroll
    for (int mi = 0; mi < 2; ++mi)
#pragma unroll
      for (int i = 0; i < 8; ++i) acc[mi][t][i] += bval;
  }

  int h = n0 >> 6;  // head index
#pragma unroll
  for (int mi = 0; mi < 2; ++mi) {
    int mbase = m0 + 16 * mi;
    if (mat < 2) {
      // RoPE in-register: partner column d+-32 lives in accumulator t+-2,
      // same lane (col) and same element (row).
      float nacc[4][8];
#pragma unroll
      for (int t = 0; t < 4; ++t) {
        int d = 16 * t + lm;
        int j = d & 31;
        float inv = __expf(-(float)j * 0.2878231366f);  // ln(10000)/32
#pragma unroll
        for (int i = 0; i < 8; ++i) {
          int mrow = mbase + i + 8 * half;
          int s    = mrow & (S_LEN - 1);
          float fr = (float)s * inv, sv, cv;
          __sincosf(fr, &sv, &cv);
          float part = (d < 32) ? acc[mi][t + 2][i] : acc[mi][t - 2][i];
          nacc[t][i] = (d < 32) ? (acc[mi][t][i] * cv - part * sv)
                                : (acc[mi][t][i] * cv + part * sv);
        }
      }
      unsigned short* dst = (mat == 0) ? qb : kb;
#pragma unroll
      for (int t = 0; t < 4; ++t) {
        int d = 16 * t + lm;
#pragma unroll
        for (int i = 0; i < 8; ++i) {
          int mrow = mbase + i + 8 * half;
          int b = mrow >> 11, s = mrow & (S_LEN - 1);
          dst[((((size_t)(b * H_NUM + h)) * S_LEN + s) << 6) + d] =
              f32_bf16(nacc[t][i]);
        }
      }
    } else {
#pragma unroll
      for (int t = 0; t < 4; ++t) {
        int d = 16 * t + lm;
#pragma unroll
        for (int i = 0; i < 8; ++i) {
          int mrow = mbase + i + 8 * half;
          int b = mrow >> 11, s = mrow & (S_LEN - 1);
          vb[((size_t)(b * H_NUM + h) * D_DIM + d) * S_LEN + s] =
              f32_bf16(acc[mi][t][i]);
        }
      }
    }
  }
}

// ---------------- kernel 4: flash attention --------------------------------
// One wave: 16 query rows of one (batch, head). 32-key blocks:
// 4 QK^T WMMAs + online softmax + LDS C->A relayout + 4 PV WMMAs.
__global__ void __launch_bounds__(256)
k_attn(const unsigned short* __restrict__ qb, const unsigned short* __restrict__ kb,
       const unsigned short* __restrict__ vb, const float* __restrict__ mask,
       float* __restrict__ out) {
  __shared__ unsigned short ldsP[8][16 * 32];  // per-wave P staging (C->A layout)

  int lane = threadIdx.x & 31;
  int wid  = threadIdx.x >> 5;
  int widG = blockIdx.x * 8 + wid;
  int b    = widG >> 11;
  int rem  = widG & 2047;
  int h    = rem >> 7;
  int q0   = (rem & 127) << 4;

  size_t headOff = (size_t)(b * H_NUM + h) * S_LEN * D_DIM;
  const unsigned short* qh = qb + headOff;
  const unsigned short* kh = kb + headOff;
  const unsigned short* vh = vb + headOff;          // (D, S) layout
  const float* mrow = mask + (size_t)b * S_LEN;

  int half = lane >> 4, lm = lane & 15;

  v16bf qA0 = load_frag(qh, D_DIM, q0, 0, lane);
  v16bf qA1 = load_frag(qh, D_DIM, q0, 32, lane);

  v8f o[4];
#pragma unroll
  for (int t = 0; t < 4; ++t) o[t] = (v8f)(0.0f);
  float mrun[8], lrun[8];
#pragma unroll
  for (int i = 0; i < 8; ++i) { mrun[i] = -3.0e38f; lrun[i] = 0.0f; }

  for (int j0 = 0; j0 < S_LEN; j0 += 32) {
    v8f s0 = (v8f)(0.0f), s1 = (v8f)(0.0f);
    {
      v16bf kB;
      kB = load_frag(kh, D_DIM, j0, 0, lane);       s0 = wmma_bf16(qA0, kB, s0);
      kB = load_frag(kh, D_DIM, j0, 32, lane);      s0 = wmma_bf16(qA1, kB, s0);
      kB = load_frag(kh, D_DIM, j0 + 16, 0, lane);  s1 = wmma_bf16(qA0, kB, s1);
      kB = load_frag(kh, D_DIM, j0 + 16, 32, lane); s1 = wmma_bf16(qA1, kB, s1);
    }
    float mk0 = mrow[j0 + lm], mk1 = mrow[j0 + 16 + lm];

#pragma unroll
    for (int i = 0; i < 8; ++i) {
      float a0 = s0[i] * 0.125f + mk0;    // 1/sqrt(64) + mask
      float a1 = s1[i] * 0.125f + mk1;
      float mx = fmaxf(a0, a1);
#pragma unroll
      for (int sh = 1; sh < 16; sh <<= 1)           // row reduce (16-lane group)
        mx = fmaxf(mx, __shfl_xor(mx, sh, 32));
      float mnew  = fmaxf(mrun[i], mx);
      float scale = __expf(mrun[i] - mnew);
      float p0 = __expf(a0 - mnew);
      float p1 = __expf(a1 - mnew);
      float rs = p0 + p1;
#pragma unroll
      for (int sh = 1; sh < 16; sh <<= 1)
        rs += __shfl_xor(rs, sh, 32);
      lrun[i] = lrun[i] * scale + rs;
      mrun[i] = mnew;
#pragma unroll
      for (int t = 0; t < 4; ++t) o[t][i] *= scale;
      int row = i + 8 * half;
      ldsP[wid][row * 32 + lm]      = f32_bf16(p0);
      ldsP[wid][row * 32 + 16 + lm] = f32_bf16(p1);
    }

    // Wave-private LDS round-trip: stores (C layout) -> loads (A layout).
    asm volatile("s_wait_dscnt 0x0" ::: "memory");

    v16bf pA;
    {
      const unsigned short* lp = &ldsP[wid][lm * 32 + 8 * half];
      FragU f;
      f.q[0] = *(const uint4*)(lp);
      f.q[1] = *(const uint4*)(lp + 16);
      pA = f.v;
    }

#pragma unroll
    for (int t = 0; t < 4; ++t) {
      v16bf vB = load_frag(vh, S_LEN, 16 * t, j0, lane);  // (D,S): cols=d, K=keys
      o[t] = wmma_bf16(pA, vB, o[t]);
    }
  }

#pragma unroll
  for (int t = 0; t < 4; ++t) {
    int d = 16 * t + lm;
#pragma unroll
    for (int i = 0; i < 8; ++i) {
      int qrow = q0 + i + 8 * half;
      out[(size_t)(b * S_LEN + qrow) * E_DIM + h * D_DIM + d] = o[t][i] / lrun[i];
    }
  }
}

// ---------------------------------------------------------------------------
extern "C" void kernel_launch(void* const* d_in, const int* in_sizes, int n_in,
                              void* d_out, int out_size, void* d_ws, size_t ws_size,
                              hipStream_t stream) {
  const float* x    = (const float*)d_in[0];
  const float* mask = (const float*)d_in[1];
  const float* Wq   = (const float*)d_in[2];
  const float* bq   = (const float*)d_in[3];
  const float* Wk   = (const float*)d_in[4];
  const float* bk   = (const float*)d_in[5];
  const float* Wv   = (const float*)d_in[6];
  const float* bv   = (const float*)d_in[7];
  float* out = (float*)d_out;

  char* ws = (char*)d_ws;
  unsigned short* xb = (unsigned short*)ws;                                // 8 MB
  unsigned short* wt = (unsigned short*)(ws + (size_t)8  * 1024 * 1024);   // 6 MB
  unsigned short* qb = (unsigned short*)(ws + (size_t)14 * 1024 * 1024);   // 8 MB
  unsigned short* kb = (unsigned short*)(ws + (size_t)22 * 1024 * 1024);   // 8 MB
  unsigned short* vb = (unsigned short*)(ws + (size_t)30 * 1024 * 1024);   // 8 MB

  int n4 = (M_ROWS * E_DIM) / 4;
  k_cvt_x<<<n4 / 256, 256, 0, stream>>>((const float4*)x, (ushort4*)xb, n4);
  k_cvt_wt<<<(3 * 1024 * 1024) / 256, 256, 0, stream>>>(Wq, Wk, Wv, wt);
  k_qkv<<<768, 256, 0, stream>>>(xb, wt, bq, bk, bv, qb, kb, vb);
  k_attn<<<512, 256, 0, stream>>>(qb, kb, vb, mask, out);
}
